// ACD_VAE_A2C_58600533787130
// MI455X (gfx1250) — matrix operand order repaired and checked by reading, not compiled, not run on hardware
//
#include <hip/hip_runtime.h>
#include <hip/hip_bf16.h>

typedef _Float16 half_t;
typedef __attribute__((ext_vector_type(16))) _Float16 v16h;
typedef __attribute__((ext_vector_type(8)))  _Float16 v8h;
typedef __attribute__((ext_vector_type(8)))  float    v8f;
typedef __attribute__((ext_vector_type(4)))  float    v4f;

#define DIVUP(a,b) (((a)+(b)-1)/(b))

// Problem constants
#define Bsz   128
#define Nag   16
#define OBSD  64
#define Hd    256
#define Md    32        // 2*N nodes
#define Ed    992       // M*(M-1) edges
#define NOUTD 65
#define REROWS (Bsz*Ed) // 126976 edge rows

// ---------------------------------------------------------------------------
// f32 -> f16 weight conversion
// ---------------------------------------------------------------------------
__global__ void k_cvt(const float* __restrict__ s, half_t* __restrict__ d, int n) {
  int i = blockIdx.x * 256 + threadIdx.x;
  if (i < n) d[i] = (half_t)s[i];
}

__device__ __forceinline__ const float* nodePtr(const float* p0, const float* p1,
                                                int b, int node, int dn) {
  if (p1) return (node < 16 ? p0 : p1) + (size_t)(b * 16 + (node & 15)) * dn;
  return p0 + (size_t)(b * Md + node) * dn;
}

// ---------------------------------------------------------------------------
// WMMA GEMM: out[R x Ncols] = feederA[R x K] @ W^T  (W is [Ncols x K] f16)
//
// A-row feeder (regions concatenated along K; widths are kernel scalars so
// region selection is wave-uniform -> s_cbranch, no exec masking):
//   mode 0: dense row-major (lda)
//   mode 1: node rows (b,m): hobs/hrew pair feature ++ pe extra (stride ldpe)
//   mode 2: edge rows (b,e): send ++ recv ++ pe extra
//           recv = e/31, send = (e%31) + (e%31 >= recv)
// Out-of-range rows/cols are clamped (not zeroed) and masked at the store;
// K-tail is zeroed on the B side (scalar condition).
//
// Block: 256 threads (8 wave32). Block tile 128x64, K-chunk 32.
// Each wave: 32x32 out tile = 2x2 v_wmma_f32_16x16x32_f16, fp32 accumulate.
// ---------------------------------------------------------------------------
#define AS_STRIDE 40   // 32 halves + pad (80B rows -> 16B-aligned runs)

__global__ void __launch_bounds__(256) k_gemm_wmma(
    const float* __restrict__ p0, const float* __restrict__ p1,
    const float* __restrict__ pe,
    int mode, int lda, int dn, int de, int ldpe,
    const half_t* __restrict__ W, int K, int Ncols, int R,
    const float* __restrict__ bias, int relu,
    const float* __restrict__ rel, int relidx, int accum,
    int outmap, int outbase, int ldo, float* __restrict__ out)
{
  __shared__ half_t As[128][AS_STRIDE];
  __shared__ half_t Bs[64][AS_STRIDE];

  const int tid      = threadIdx.x;
  const int rowBlock = blockIdx.y * 128;
  const int colBlock = blockIdx.x * 64;

  // ---- scalar region widths ----------------------------------------------
  int reg0, reg1, reg2;
  if (mode == 0)      { reg0 = lda; reg1 = 0;  reg2 = 0;  }
  else if (mode == 1) { reg0 = dn;  reg1 = 0;  reg2 = de; }
  else                { reg0 = dn;  reg1 = dn; reg2 = de; }
  const int reg01  = reg0 + reg1;
  const int reg012 = reg01 + reg2;

  // ---- per-thread A row pointers (row clamped to R-1) ---------------------
  const int arow = tid & 127;
  int row = rowBlock + arow; if (row >= R) row = R - 1;
  const float* rowA = nullptr; const float* rowB = nullptr; const float* rowC = nullptr;
  if (mode == 0) {
    rowA = p0 + (size_t)row * lda;
  } else if (mode == 1) {
    int b = row >> 5, m = row & 31;
    rowA = nodePtr(p0, p1, b, m, dn);
    if (de) rowC = pe + (size_t)row * ldpe;
  } else {
    unsigned ur = (unsigned)row;
    unsigned b = ur / (unsigned)Ed;
    unsigned e = ur - b * (unsigned)Ed;
    unsigned recv = e / 31u;
    unsigned j = e - recv * 31u;
    unsigned send = j + (j >= recv ? 1u : 0u);
    rowA = nodePtr(p0, p1, (int)b, (int)send, dn);
    rowB = nodePtr(p0, p1, (int)b, (int)recv, dn);
    if (de) rowC = pe + (size_t)row * ldpe;
  }
  // wave-uniform staging coordinates -> SGPRs
  const int tidU  = __builtin_amdgcn_readfirstlane(tid);
  const int ahalf = (tidU >> 7) & 1;
  const int bq    = (tidU >> 6) & 3;

  // ---- B staging (column clamped to Ncols-1) ------------------------------
  const int bcol = tid & 63;
  int wn = colBlock + bcol; if (wn >= Ncols) wn = Ncols - 1;
  const half_t* wrow = W + (size_t)wn * K;
  const bool k8 = ((K & 7) == 0);

  const int wid  = tidU >> 5;
  const int lane = tid & 31;
  const int wr   = (wid >> 1) * 32;   // wave row tile in block
  const int wc   = (wid & 1) * 32;    // wave col tile in block
  const int lrow = lane & 15;
  const int hi   = lane >> 4;         // lane half
  const int kbA  = hi * 8;            // A frag: runs [kbA..+7] and [16+kbA..+7]
  const int kbB  = hi * 8;            // B frag: runs [kbB..+7] and [16+kbB..+7]

  v8f acc00 = {}; v8f acc01 = {}; v8f acc10 = {}; v8f acc11 = {};

  for (int k0 = 0; k0 < K; k0 += 32) {
    __syncthreads();
    // ---- stage A tile (f32 -> f16); region choice is scalar ---------------
    {
      const int kbeg = k0 + ahalf * 16;   // wave-uniform
      half_t tmp[16];
      const float* src = nullptr;
      if (kbeg + 16 <= reg0)                            src = rowA + kbeg;
      else if (kbeg >= reg0  && kbeg + 16 <= reg01)     src = rowB + (kbeg - reg0);
      else if (kbeg >= reg01 && kbeg + 16 <= reg012)    src = rowC + (kbeg - reg01);
      if (src) {
        const v4f* s4 = (const v4f*)src;
        #pragma unroll
        for (int q = 0; q < 4; ++q) {
          v4f v = s4[q];
          tmp[q * 4 + 0] = (half_t)v[0];
          tmp[q * 4 + 1] = (half_t)v[1];
          tmp[q * 4 + 2] = (half_t)v[2];
          tmp[q * 4 + 3] = (half_t)v[3];
        }
      } else {
        #pragma unroll
        for (int jj = 0; jj < 16; ++jj) {
          int k = kbeg + jj;              // scalar per unrolled step
          float v = 0.f;
          if (k < reg0)        v = rowA[k];
          else if (k < reg01)  v = rowB[k - reg0];
          else if (k < reg012) v = rowC[k - reg01];
          tmp[jj] = (half_t)v;
        }
      }
      v8h t0, t1;
      #pragma unroll
      for (int jj = 0; jj < 8; ++jj) { t0[jj] = tmp[jj]; t1[jj] = tmp[8 + jj]; }
      v8h* dst = (v8h*)&As[arow][ahalf * 16];
      dst[0] = t0; dst[1] = t1;
    }
    // ---- stage B tile from f16 weights (scalar conditions) ----------------
    {
      const int kw = k0 + bq * 8;         // wave-uniform
      v8h bv;
      if (k8 && (kw + 8 <= K)) {
        bv = *(const v8h*)(wrow + kw);
      } else {
        #pragma unroll
        for (int jj = 0; jj < 8; ++jj)
          bv[jj] = (kw + jj < K) ? wrow[kw + jj] : (half_t)0.f;
      }
      *(v8h*)&Bs[bcol][bq * 8] = bv;
      if (k0 + 32 < K)
        __builtin_prefetch(wrow + k0 + 32, 0, 3);
    }
    __syncthreads();

    // ---- fragments from LDS (16B-aligned b128 reads) -----------------------
    const half_t* ap0 = &As[wr + lrow][0];
    const half_t* ap1 = &As[wr + 16 + lrow][0];
    const half_t* bp0 = &Bs[wc + lrow][0];
    const half_t* bp1 = &Bs[wc + 16 + lrow][0];
    v8h a0l = *(const v8h*)(ap0 + kbA);
    v8h a0h = *(const v8h*)(ap0 + 16 + kbA);
    v8h a1l = *(const v8h*)(ap1 + kbA);
    v8h a1h = *(const v8h*)(ap1 + 16 + kbA);
    v8h b0l = *(const v8h*)(bp0 + kbB);
    v8h b0h = *(const v8h*)(bp0 + kbB + 16);
    v8h b1l = *(const v8h*)(bp1 + kbB);
    v8h b1h = *(const v8h*)(bp1 + kbB + 16);
    v16h a0 = __builtin_shufflevector(a0l, a0h, 0,1,2,3,4,5,6,7,8,9,10,11,12,13,14,15);
    v16h a1 = __builtin_shufflevector(a1l, a1h, 0,1,2,3,4,5,6,7,8,9,10,11,12,13,14,15);
    v16h b0 = __builtin_shufflevector(b0l, b0h, 0,1,2,3,4,5,6,7,8,9,10,11,12,13,14,15);
    v16h b1 = __builtin_shufflevector(b1l, b1h, 0,1,2,3,4,5,6,7,8,9,10,11,12,13,14,15);

    acc00 = __builtin_amdgcn_wmma_f32_16x16x32_f16(false, a0, false, b0, (short)0, acc00, false, false);
    acc01 = __builtin_amdgcn_wmma_f32_16x16x32_f16(false, a0, false, b1, (short)0, acc01, false, false);
    acc10 = __builtin_amdgcn_wmma_f32_16x16x32_f16(false, a1, false, b0, (short)0, acc10, false, false);
    acc11 = __builtin_amdgcn_wmma_f32_16x16x32_f16(false, a1, false, b1, (short)0, acc11, false, false);
  }

  // ---- epilogue ------------------------------------------------------------
  v8f accs[2][2] = { { acc00, acc01 }, { acc10, acc11 } };
  #pragma unroll
  for (int ti = 0; ti < 2; ++ti) {
    #pragma unroll
    for (int tj = 0; tj < 2; ++tj) {
      int cg = colBlock + wc + tj * 16 + lrow;
      if (cg >= Ncols) continue;
      float bv = bias ? bias[cg] : 0.f;
      #pragma unroll
      for (int v = 0; v < 8; ++v) {
        int rg = rowBlock + wr + ti * 16 + hi * 8 + v;
        if (rg >= R) continue;
        float val = accs[ti][tj][v] + bv;
        if (relu) val = fmaxf(val, 0.f);
        if (rel)  val *= rel[(size_t)rg * 2 + relidx];
        int ro = (outmap == 1) ? (((rg >> 4) << 5) + (rg & 15) + outbase * 16) : rg;
        size_t oi = (size_t)ro * ldo + cg;
        if (accum) out[oi] += val; else out[oi] = val;
      }
    }
  }
}

// ---------------------------------------------------------------------------
// GRU gate fusion: h' = (1-z)*tanh(in + r*hn) + z*h
// ---------------------------------------------------------------------------
__global__ void k_gru_gate(const float* __restrict__ gi, const float* __restrict__ gh,
                           const float* __restrict__ h, float* __restrict__ hout,
                           int total) {
  int i = blockIdx.x * 256 + threadIdx.x;
  if (i >= total) return;
  int r = i >> 8, c = i & 255;
  size_t b = (size_t)r * 768;
  float ir = gi[b + c],       hr = gh[b + c];
  float iz = gi[b + 256 + c], hz = gh[b + 256 + c];
  float in = gi[b + 512 + c], hn = gh[b + 512 + c];
  float rg = 1.f / (1.f + __expf(-(ir + hr)));
  float zg = 1.f / (1.f + __expf(-(iz + hz)));
  float ng = tanhf(in + rg * hn);
  size_t hidx = (size_t)r * 256 + c;
  hout[hidx] = (1.f - zg) * ng + zg * h[hidx];
}

// gi for reward GRUs (K=1): gi[r,c] = rew[r]*Wih[c] + bih[c]
__global__ void k_rew_gi(const float* __restrict__ rew, const float* __restrict__ wih,
                         const float* __restrict__ bih, float* __restrict__ gi) {
  int i = blockIdx.x * 256 + threadIdx.x;
  if (i >= 2048 * 768) return;
  int r = i / 768, c = i - r * 768;
  gi[i] = rew[r] * wih[c] + bih[c];
}

// edge2node_sum: RECV[e]=e/31 -> contiguous 31-row sum per node, * scale
__global__ void k_sum31(const float* __restrict__ src, float* __restrict__ dst,
                        int d, int ldo, float scale) {
  int i = blockIdx.x * 256 + threadIdx.x;
  int total = Bsz * Md * d;
  if (i >= total) return;
  int c  = i % d;
  int ni = i / d;              // b*32 + m
  int b  = ni >> 5, m = ni & 31;
  const float* s = src + ((size_t)b * Ed + (size_t)m * 31) * d + c;
  float acc = 0.f;
  #pragma unroll 1
  for (int t = 0; t < 31; ++t) acc += s[(size_t)t * d];
  dst[(size_t)ni * ldo + c] = acc * scale;
}

// fc_out (256 -> 2) + softmax over ET=2
__global__ void k_fcout_softmax(const float* __restrict__ x, const float* __restrict__ W,
                                const float* __restrict__ bb, float* __restrict__ rel,
                                int R) {
  int r = blockIdx.x * 256 + threadIdx.x;
  if (r >= R) return;
  const float* xr = x + (size_t)r * 256;
  float a0 = bb[0], a1 = bb[1];
  for (int i = 0; i < 256; i += 4) {
    v4f v = *(const v4f*)(xr + i);
    v4f w0 = *(const v4f*)(W + i);
    v4f w1 = *(const v4f*)(W + 256 + i);
    a0 += v[0]*w0[0] + v[1]*w0[1] + v[2]*w0[2] + v[3]*w0[3];
    a1 += v[0]*w1[0] + v[1]*w1[1] + v[2]*w1[2] + v[3]*w1[3];
  }
  float m  = fmaxf(a0, a1);
  float e0 = __expf(a0 - m), e1 = __expf(a1 - m);
  float s  = e0 + e1;
  rel[(size_t)r * 2]     = e0 / s;
  rel[(size_t)r * 2 + 1] = e1 / s;
}

// message_obs[b,t,d] = sum_{s!=t} rel[b, e(recv=s,send=t), 1] * msgs[b,s,d]
__global__ void k_msg_einsum(const float* __restrict__ rel, const float* __restrict__ msgs,
                             float* __restrict__ mo) {
  int i = blockIdx.x * 256 + threadIdx.x;
  if (i >= Bsz * Nag * OBSD) return;
  int d = i & 63;
  int t = (i >> 6) & 15;
  int b = i >> 10;
  float acc = 0.f;
  for (int s = 0; s < Md; ++s) {
    if (s == t) continue;
    int e = s * 31 + (t - (t > s ? 1 : 0));
    float w = rel[((size_t)b * Ed + e) * 2 + 1];
    acc += w * msgs[((size_t)b * Md + s) * 64 + d];
  }
  mo[i] = acc;
}

// per-agent head MLP: relu(x @ W1[n] + b1[n]) @ W2[n] + b2[n]
__global__ void __launch_bounds__(128) k_headmlp(
    const float* __restrict__ obs, const float* __restrict__ msgobs,
    const float* __restrict__ W1, const float* __restrict__ b1,
    const float* __restrict__ W2, const float* __restrict__ b2,
    float* __restrict__ outp, int dout)
{
  __shared__ float xin[128];
  __shared__ float hid[128];
  int bn = blockIdx.x;         // b*16 + n
  int n  = bn & 15;
  int t  = threadIdx.x;
  xin[t] = (t < 64) ? obs[(size_t)bn * 64 + t] : msgobs[(size_t)bn * 64 + (t - 64)];
  __syncthreads();
  const float* w1 = W1 + (size_t)n * 128 * 128;
  float acc = b1[n * 128 + t];
  for (int i = 0; i < 128; ++i) acc += xin[i] * w1[i * 128 + t];
  hid[t] = fmaxf(acc, 0.f);
  __syncthreads();
  if (t < dout) {
    const float* w2 = W2 + (size_t)n * 128 * dout;
    float o = b2[n * dout + t];
    for (int i = 0; i < 128; ++i) o += hid[i] * w2[i * dout + t];
    outp[(size_t)bn * dout + t] = o;
  }
}

// ---------------------------------------------------------------------------
extern "C" void kernel_launch(void* const* d_in, const int* in_sizes, int n_in,
                              void* d_out, int out_size, void* d_ws, size_t ws_size,
                              hipStream_t stream) {
  (void)in_sizes; (void)n_in; (void)out_size; (void)ws_size;

  const float* obs  = (const float*)d_in[0];
  const float* rew  = (const float*)d_in[1];
  const float* phO  = (const float*)d_in[2];
  const float* phR  = (const float*)d_in[3];
  auto F = [&](int i) { return (const float*)d_in[i]; };

  // ---- output slots -------------------------------------------------------
  float* out   = (float*)d_out;
  float* oACT  = out;                 // (B,N,16)   32768
  float* oCRI  = out + 32768;         // (B,N,1)    2048
  float* oSCM  = out + 34816;         // (B,M,65)   266240
  float* oREL  = out + 301056;        // (B,992,2)  253952
  float* oHOBS = out + 555008;        // (1,B*N,H)  524288
  float* oHREW = out + 1079296;       // (1,B*N,H)  524288

  // ---- workspace layout ---------------------------------------------------
  float* fw = (float*)d_ws;
  size_t off = 0;
  auto take = [&](size_t n) { float* p = fw + off; off += (n + 63) & ~(size_t)63; return p; };
  float* GI   = take((size_t)2048 * 768);
  float* GH   = take((size_t)2048 * 768);
  float* HOBS = take((size_t)2048 * 256);   // a2c obs hidden
  float* HREW = take((size_t)2048 * 256);   // a2c rew hidden
  float* TN   = take((size_t)4096 * 256);   // node scratch
  float* X1   = take((size_t)4096 * 256);
  float* X3   = take((size_t)4096 * 256);
  float* AGG  = take((size_t)4096 * 256);
  float* AGGN = take((size_t)4096 * 80);    // 65 cols padded to 80 (16B-aligned rows)
  float* DH1  = take((size_t)4096 * 256);
  float* DH2  = take((size_t)4096 * 256);
  float* MSGS = take((size_t)4096 * 64);
  float* MOBS = take((size_t)2048 * 64);
  float* E1   = take((size_t)REROWS * 256); // edge buffer A (xskip / mlp4out / allmsgs)
  float* E2   = take((size_t)REROWS * 256); // edge buffer B (MLP hiddens)
  half_t* hbase = (half_t*)(fw + off);
  size_t hoff = 0;

  auto cvt = [&](int idx, size_t n) {
    half_t* dst = hbase + hoff; hoff += (n + 7) & ~(size_t)7;
    k_cvt<<<DIVUP(n, 256), 256, 0, stream>>>((const float*)d_in[idx], dst, (int)n);
    return dst;
  };
  // f16 weights
  half_t* wAoIH = cvt(4,  768 * 64);   half_t* wAoHH = cvt(5,  768 * 256);
  half_t* wArHH = cvt(9,  768 * 256);
  half_t* wSoIH = cvt(12, 768 * 64);   half_t* wSoHH = cvt(13, 768 * 256);
  half_t* wSrHH = cvt(17, 768 * 256);
  half_t* wE1a = cvt(20, 256 * 256);   half_t* wE1b = cvt(22, 256 * 256);
  half_t* wE2a = cvt(24, 256 * 512);   half_t* wE2b = cvt(26, 256 * 256);
  half_t* wE3a = cvt(28, 256 * 256);   half_t* wE3b = cvt(30, 256 * 256);
  half_t* wE4a = cvt(32, 256 * 768);   half_t* wE4b = cvt(34, 256 * 256);
  half_t* wDM1 = cvt(38, 2 * 256 * 512);
  half_t* wDM2 = cvt(40, 2 * 65 * 256);
  half_t* wDO1 = cvt(42, 256 * 321);
  half_t* wDO2 = cvt(44, 256 * 256);
  half_t* wDO3 = cvt(46, 65 * 256);
  half_t* wM0  = cvt(48, 64 * 256);
  half_t* wM1  = cvt(50, 64 * 256);

  auto gemm = [&](const float* p0, const float* p1, const float* pe,
                  int mode, int lda, int dn, int de, int ldpe,
                  const half_t* W, int K, int Ncols, int R,
                  const float* bias, int relu,
                  const float* rel, int relidx, int accum,
                  int outmap, int outbase, int ldo, float* o) {
    dim3 g(DIVUP(Ncols, 64), DIVUP(R, 128));
    k_gemm_wmma<<<g, 256, 0, stream>>>(p0, p1, pe, mode, lda, dn, de, ldpe,
                                       W, K, Ncols, R, bias, relu,
                                       rel, relidx, accum, outmap, outbase, ldo, o);
  };
  const int G768 = DIVUP(2048 * 768, 256);
  const int G512k = DIVUP(2048 * 256, 256);

  // ---- GRUs ---------------------------------------------------------------
  // a2c obs
  gemm(obs, nullptr, nullptr, 0, 64, 0, 0, 0, wAoIH, 64, 768, 2048, F(6), 0, nullptr, 0, 0, 0, 0, 768, GI);
  gemm(phO, nullptr, nullptr, 0, 256, 0, 0, 0, wAoHH, 256, 768, 2048, F(7), 0, nullptr, 0, 0, 0, 0, 768, GH);
  k_gru_gate<<<G512k, 256, 0, stream>>>(GI, GH, phO, HOBS, 2048 * 256);
  // a2c rew
  k_rew_gi<<<G768, 256, 0, stream>>>(rew, F(8), F(10), GI);
  gemm(phR, nullptr, nullptr, 0, 256, 0, 0, 0, wArHH, 256, 768, 2048, F(11), 0, nullptr, 0, 0, 0, 0, 768, GH);
  k_gru_gate<<<G512k, 256, 0, stream>>>(GI, GH, phR, HREW, 2048 * 256);
  // scm obs -> d_out h_out_obs
  gemm(obs, nullptr, nullptr, 0, 64, 0, 0, 0, wSoIH, 64, 768, 2048, F(14), 0, nullptr, 0, 0, 0, 0, 768, GI);
  gemm(phO, nullptr, nullptr, 0, 256, 0, 0, 0, wSoHH, 256, 768, 2048, F(15), 0, nullptr, 0, 0, 0, 0, 768, GH);
  k_gru_gate<<<G512k, 256, 0, stream>>>(GI, GH, phO, oHOBS, 2048 * 256);
  // scm rew -> d_out h_out_rew
  k_rew_gi<<<G768, 256, 0, stream>>>(rew, F(16), F(18), GI);
  gemm(phR, nullptr, nullptr, 0, 256, 0, 0, 0, wSrHH, 256, 768, 2048, F(19), 0, nullptr, 0, 0, 0, 0, 768, GH);
  k_gru_gate<<<G512k, 256, 0, stream>>>(GI, GH, phR, oHREW, 2048 * 256);

  // ---- encoder ------------------------------------------------------------
  // mlp1 over nodes (scm_hidden pair rows)
  gemm(oHOBS, oHREW, nullptr, 1, 0, 256, 0, 0, wE1a, 256, 256, 4096, F(21), 1, nullptr, 0, 0, 0, 0, 256, TN);
  gemm(TN, nullptr, nullptr, 0, 256, 0, 0, 0, wE1b, 256, 256, 4096, F(23), 0, nullptr, 0, 0, 0, 0, 256, X1);
  // mlp2 over edges: A = node2edge(X1) (fused gather)
  gemm(X1, nullptr, nullptr, 2, 0, 256, 0, 0, wE2a, 512, 256, REROWS, F(25), 1, nullptr, 0, 0, 0, 0, 256, E2);
  gemm(E2, nullptr, nullptr, 0, 256, 0, 0, 0, wE2b, 256, 256, REROWS, F(27), 0, nullptr, 0, 0, 0, 0, 256, E1); // x_skip
  // agg = edge2node_sum / (DEG+1e-6)
  k_sum31<<<DIVUP(Bsz * Md * 256, 256), 256, 0, stream>>>(E1, AGG, 256, 256, 1.f / (31.f + 1e-6f));
  // mlp3 over nodes
  gemm(AGG, nullptr, nullptr, 0, 256, 0, 0, 0, wE3a, 256, 256, 4096, F(29), 1, nullptr, 0, 0, 0, 0, 256, TN);
  gemm(TN, nullptr, nullptr, 0, 256, 0, 0, 0, wE3b, 256, 256, 4096, F(31), 0, nullptr, 0, 0, 0, 0, 256, X3);
  // mlp4 over edges: A = [node2edge(X3), x_skip]
  gemm(X3, nullptr, E1, 2, 0, 256, 256, 256, wE4a, 768, 256, REROWS, F(33), 1, nullptr, 0, 0, 0, 0, 256, E2);
  gemm(E2, nullptr, nullptr, 0, 256, 0, 0, 0, wE4b, 256, 256, REROWS, F(35), 0, nullptr, 0, 0, 0, 0, 256, E1);
  // fc_out (dout=2) + softmax -> rel_type in d_out
  k_fcout_softmax<<<DIVUP(REROWS, 256), 256, 0, stream>>>(E1, F(36), F(37), oREL, REROWS);

  // ---- decoder ------------------------------------------------------------
  for (int i = 0; i < 2; ++i) {
    const half_t* w1 = wDM1 + (size_t)i * 256 * 512;
    const half_t* w2 = wDM2 + (size_t)i * 65 * 256;
    const float*  b1 = F(39) + i * 256;
    const float*  b2 = F(41) + i * 65;
    // msg1: edge gather of scm_hidden pair, relu
    gemm(oHOBS, oHREW, nullptr, 2, 0, 256, 0, 0, w1, 512, 256, REROWS, b1, 1, nullptr, 0, 0, 0, 0, 256, E2);
    // msg2: relu, scale by rel[:,:,i]; i==0 writes, i==1 accumulates (all_msgs in E1)
    gemm(E2, nullptr, nullptr, 0, 256, 0, 0, 0, w2, 256, 65, REROWS, b2, 1, oREL, i, (i == 1), 0, 0, 65, E1);
  }
  // agg over edges -> (B,M,65) padded to stride 80
  k_sum31<<<DIVUP(Bsz * Md * 65, 256), 256, 0, stream>>>(E1, AGGN, 65, 80, 1.f);
  // dec_out1: A = [scm_hidden, agg] (K=321), relu
  gemm(oHOBS, oHREW, AGGN, 1, 0, 256, 65, 80, wDO1, 321, 256, 4096, F(43), 1, nullptr, 0, 0, 0, 0, 256, DH1);
  gemm(DH1, nullptr, nullptr, 0, 256, 0, 0, 0, wDO2, 256, 256, 4096, F(45), 1, nullptr, 0, 0, 0, 0, 256, DH2);
  gemm(DH2, nullptr, nullptr, 0, 256, 0, 0, 0, wDO3, 256, 65, 4096, F(47), 0, nullptr, 0, 0, 0, 0, 65, oSCM);

  // ---- message passing + heads (a2c hidden) -------------------------------
  gemm(HOBS, nullptr, nullptr, 0, 256, 0, 0, 0, wM0, 256, 64, 2048, F(49), 0, nullptr, 0, 0, 1, 0, 64, MSGS);
  gemm(HREW, nullptr, nullptr, 0, 256, 0, 0, 0, wM1, 256, 64, 2048, F(51), 0, nullptr, 0, 0, 1, 1, 64, MSGS);
  k_msg_einsum<<<DIVUP(Bsz * Nag * OBSD, 256), 256, 0, stream>>>(oREL, MSGS, MOBS);
  k_headmlp<<<Bsz * Nag, 128, 0, stream>>>(obs, MOBS, F(52), F(53), F(54), F(55), oACT, 16);
  k_headmlp<<<Bsz * Nag, 128, 0, stream>>>(obs, MOBS, F(56), F(57), F(58), F(59), oCRI, 1);
}
